// BrainGCN_68436008894831
// MI455X (gfx1250) — matrix operand order repaired and checked by reading, not compiled, not run on hardware
//
#include <hip/hip_runtime.h>
#include <hip/hip_bf16.h>
#include <math.h>

// ---------------------------------------------------------------------------
// BrainGCN on MI455X (gfx1250, wave32)
//   conv1: h1 = tanh(scatter(norm * (x@W1)) + b1)
//   conv2: h2 = tanh(scatter(norm * (h1@W2)) + b2)
//   head : out = tanh(h2@fw1 + fb1) @ fw2 + fb2
// GEMMs: v_wmma_f32_16x16x32_bf16, operands pre-converted/pre-swizzled to bf16
// so the inner loop is pure b128 loads + WMMA.  Edge aggregation is wave-per-
// edge float2 gather + global f32 atomics (L2-resident: 12.8MB features vs
// 192MB L2).
// ---------------------------------------------------------------------------

typedef __attribute__((ext_vector_type(16))) __bf16        v16bf;
typedef __attribute__((ext_vector_type(8)))  float         v8f;
typedef __attribute__((ext_vector_type(4)))  unsigned int  u32x4;

union AV {
  v16bf v;
  u32x4 q[2];
};

__device__ __forceinline__ unsigned short f2bf_rne(float f) {
  union { float f; unsigned u; } c;
  c.f = f;
  unsigned u = c.u;
  u += 0x7fffu + ((u >> 16) & 1u);      // round-to-nearest-even into bf16
  return (unsigned short)(u >> 16);
}

// ---------------------------------------------------------------------------
// Degree / normalization.  deg[i] = (# edges with dst==i) + 1 (self loop)
// ---------------------------------------------------------------------------
__global__ void deg_init_kernel(float* __restrict__ deg, int n) {
  int i = blockIdx.x * blockDim.x + threadIdx.x;
  if (i < n) deg[i] = 1.0f;
}

__global__ void deg_count_kernel(const int* __restrict__ dst, float* __restrict__ deg, int e) {
  int i = blockIdx.x * blockDim.x + threadIdx.x;
  if (i < e) atomicAdd(&deg[dst[i]], 1.0f);
}

__global__ void deg_rsqrt_kernel(float* __restrict__ deg, int n) {
  int i = blockIdx.x * blockDim.x + threadIdx.x;
  if (i < n) deg[i] = rsqrtf(deg[i]);
}

// ---------------------------------------------------------------------------
// Streaming f32 -> bf16 (4 elements/thread; n % 4 == 0)
// ---------------------------------------------------------------------------
__global__ void f32_to_bf16_kernel(const float* __restrict__ in,
                                   unsigned short* __restrict__ out, int n4) {
  int i = blockIdx.x * blockDim.x + threadIdx.x;
  if (i < n4) {
    float4 v = ((const float4*)in)[i];
    ushort4 o;
    o.x = f2bf_rne(v.x); o.y = f2bf_rne(v.y);
    o.z = f2bf_rne(v.z); o.w = f2bf_rne(v.w);
    ((ushort4*)out)[i] = o;
  }
}

// tanh(f32) -> bf16 (conv layer activation fused with GEMM-input conversion)
__global__ void tanh_to_bf16_kernel(const float* __restrict__ in,
                                    unsigned short* __restrict__ out, int n4) {
  int i = blockIdx.x * blockDim.x + threadIdx.x;
  if (i < n4) {
    float4 v = ((const float4*)in)[i];
    ushort4 o;
    o.x = f2bf_rne(tanhf(v.x)); o.y = f2bf_rne(tanhf(v.y));
    o.z = f2bf_rne(tanhf(v.z)); o.w = f2bf_rne(tanhf(v.w));
    ((ushort4*)out)[i] = o;
  }
}

// ---------------------------------------------------------------------------
// Swizzle weight W[K][64] f32 into WMMA-B bf16 layout:
//   Bs[((kb*64 + n)*2 + g)*16 + t]  holds  W[(kb*32 + 16*g + t)][n]
// so each lane's 16-element B operand is 32 contiguous bytes (2x b128 loads).
// ---------------------------------------------------------------------------
__global__ void weight_swizzle_kernel(const float* __restrict__ W,
                                      unsigned short* __restrict__ Bs, int total /* K*64 */) {
  int idx = blockIdx.x * blockDim.x + threadIdx.x;
  if (idx < total) {
    int t  = idx & 15;
    int g  = (idx >> 4) & 1;
    int n  = (idx >> 5) & 63;
    int kb = idx >> 11;
    int k  = kb * 32 + g * 16 + t;
    Bs[idx] = f2bf_rne(W[(size_t)k * 64 + n]);
  }
}

// ---------------------------------------------------------------------------
// C[M x 64] = A[M x K] @ W[K x 64]   (A bf16 row-major, Bs pre-swizzled bf16)
// One wave per 16-row tile; the wave owns all 4 column tiles (4 accumulators,
// 4 WMMAs per 32-wide k-block).  Fully unrolled; K in {64,128}.
// ---------------------------------------------------------------------------
template <int K>
__global__ void __launch_bounds__(256)
gemm_bf16_wmma_kernel(const unsigned short* __restrict__ A,
                      const unsigned short* __restrict__ Bs,
                      float* __restrict__ C, int m_tiles) {
  const int lane = threadIdx.x & 31;
  const int wv   = threadIdx.x >> 5;
  const int tile = blockIdx.x * 8 + wv;          // 8 waves per block
  if (tile >= m_tiles) return;                    // wave-uniform exit
  const int g  = lane >> 4;                       // lane half: selects K sub-range
  const int mr = lane & 15;                       // row within tile / column within n-tile

  v8f acc[4] = {{0.f,0.f,0.f,0.f,0.f,0.f,0.f,0.f},
                {0.f,0.f,0.f,0.f,0.f,0.f,0.f,0.f},
                {0.f,0.f,0.f,0.f,0.f,0.f,0.f,0.f},
                {0.f,0.f,0.f,0.f,0.f,0.f,0.f,0.f}};

  const unsigned short* Arow = A + ((size_t)tile * 16 + mr) * K;

#pragma unroll
  for (int kb = 0; kb < K / 32; ++kb) {
    // A 16x32 bf16 operand: lane half g needs K = kb*32 + {g*16..g*16+15}
    AV a;
    a.q[0] = *(const u32x4*)(Arow + kb * 32 + g * 16);
    a.q[1] = *(const u32x4*)(Arow + kb * 32 + g * 16 + 8);
    const unsigned short* Bblk = Bs + (size_t)kb * 2048;   // 64 cols * 2 halves * 16
#pragma unroll
    for (int nt = 0; nt < 4; ++nt) {
      AV b;
      const unsigned short* bp = Bblk + (((nt * 16 + mr) * 2 + g) << 4);
      b.q[0] = *(const u32x4*)(bp);
      b.q[1] = *(const u32x4*)(bp + 8);
      acc[nt] = __builtin_amdgcn_wmma_f32_16x16x32_bf16(false, a.v, false, b.v,
                                                        (short)0, acc[nt], false, false);
    }
  }

  // C/D 16x16 f32 layout: VGPR r -> row r (lanes 0-15) / row 8+r (lanes 16-31)
  float* Crow = C + ((size_t)tile * 16 + (g << 3)) * 64 + mr;
#pragma unroll
  for (int nt = 0; nt < 4; ++nt)
#pragma unroll
    for (int r = 0; r < 8; ++r)
      Crow[(size_t)r * 64 + nt * 16] = acc[nt][r];
}

// ---------------------------------------------------------------------------
// acc[i,c] = bias[c] + dinv[i]^2 * h[i,c]   (bias + self-loop message)
// ---------------------------------------------------------------------------
__global__ void acc_init_kernel(const float* __restrict__ h, const float* __restrict__ dinv,
                                const float* __restrict__ bias, float* __restrict__ acc,
                                int n_nodes) {
  int idx = blockIdx.x * blockDim.x + threadIdx.x;
  if (idx < n_nodes * 64) {
    int node = idx >> 6;
    int c    = idx & 63;
    float w  = dinv[node];
    acc[idx] = bias[c] + w * w * h[idx];
  }
}

// ---------------------------------------------------------------------------
// One wave32 per edge: lane L handles channels 2L, 2L+1.
// acc[dst, :] += dinv[src]*dinv[dst] * h[src, :]
// ---------------------------------------------------------------------------
__global__ void scatter_edges_kernel(const int* __restrict__ src, const int* __restrict__ dst,
                                     const float* __restrict__ dinv,
                                     const float* __restrict__ h, float* __restrict__ acc,
                                     int n_edges) {
  int gid  = blockIdx.x * blockDim.x + threadIdx.x;
  int edge = gid >> 5;
  int lane = gid & 31;
  if (edge >= n_edges) return;
  int s = src[edge];
  int d = dst[edge];
  float w = dinv[s] * dinv[d];
  const float2 v = *(const float2*)(h + (size_t)s * 64 + 2 * lane);
  float* ap = acc + (size_t)d * 64 + 2 * lane;
  atomicAdd(ap,     v.x * w);
  atomicAdd(ap + 1, v.y * w);
}

// ---------------------------------------------------------------------------
// Head: out[i] = tanh(tanh(acc[i,:]) @ fw1 + fb1) @ fw2 + fb2   (node/thread)
// ---------------------------------------------------------------------------
__global__ void mlp_head_kernel(const float* __restrict__ acc,
                                const float* __restrict__ fw1, const float* __restrict__ fb1,
                                const float* __restrict__ fw2, const float* __restrict__ fb2,
                                float* __restrict__ out, int n_nodes) {
  __shared__ float sw1[64 * 32];
  __shared__ float sw2[32];
  __shared__ float sb1[32];
  for (int i = threadIdx.x; i < 64 * 32; i += blockDim.x) sw1[i] = fw1[i];
  if (threadIdx.x < 32) {
    sw2[threadIdx.x] = fw2[threadIdx.x];
    sb1[threadIdx.x] = fb1[threadIdx.x];
  }
  __syncthreads();

  int node = blockIdx.x * blockDim.x + threadIdx.x;
  if (node >= n_nodes) return;

  float xv[64];
  const float* a = acc + (size_t)node * 64;
#pragma unroll
  for (int c = 0; c < 64; ++c) xv[c] = tanhf(a[c]);

  float o = fb2[0];
#pragma unroll 4
  for (int j = 0; j < 32; ++j) {
    float hsum = sb1[j];
#pragma unroll
    for (int c = 0; c < 64; ++c) hsum += xv[c] * sw1[c * 32 + j];
    o += tanhf(hsum) * sw2[j];
  }
  out[node] = o;
}

// ---------------------------------------------------------------------------
extern "C" void kernel_launch(void* const* d_in, const int* in_sizes, int n_in,
                              void* d_out, int out_size, void* d_ws, size_t ws_size,
                              hipStream_t stream) {
  const float* x   = (const float*)d_in[0];           // [N,128]
  const int*   ei  = (const int*)d_in[1];             // [2,E]
  const float* W1  = (const float*)d_in[2];           // [128,64]
  const float* b1  = (const float*)d_in[3];           // [64]
  const float* W2  = (const float*)d_in[4];           // [64,64]
  const float* b2  = (const float*)d_in[5];           // [64]
  const float* fw1 = (const float*)d_in[6];           // [64,32]
  const float* fb1 = (const float*)d_in[7];           // [32]
  const float* fw2 = (const float*)d_in[8];           // [32,1]
  const float* fb2 = (const float*)d_in[9];           // [1]
  float* out = (float*)d_out;

  const int N = in_sizes[0] / 128;                    // 50000, divisible by 16
  const int E = in_sizes[1] / 2;                      // 800000
  const int* src = ei;
  const int* dst = ei + E;

  // ---- workspace carve (~32.2 MB) --------------------------------------
  // dinv[N] f32 | buf1[N*64] f32 | buf2[N*64] f32 (aliases xbf bf16[N*128])
  // | hbf bf16[N*64] | wbf1 bf16[128*64] | wbf2 bf16[64*64]
  char* p = (char*)d_ws;
  float* dinv = (float*)p;                 p += (size_t)N * sizeof(float);
  float* buf1 = (float*)p;                 p += (size_t)N * 64 * sizeof(float);
  float* buf2 = (float*)p;                 p += (size_t)N * 64 * sizeof(float);
  unsigned short* xbf  = (unsigned short*)buf2;        // alias: dead before acc_init1
  unsigned short* hbf  = (unsigned short*)p;  p += (size_t)N * 64 * sizeof(unsigned short);
  unsigned short* wbf1 = (unsigned short*)p;  p += 128 * 64 * sizeof(unsigned short);
  unsigned short* wbf2 = (unsigned short*)p;  p += 64 * 64 * sizeof(unsigned short);

  const int T = 256;
  const int m_tiles = N / 16;
  const int gGemm = (m_tiles + 7) / 8;                // 8 waves (tiles) per block
  const int gN    = (N + T - 1) / T;
  const int gE    = (E + T - 1) / T;
  const int gNC   = (N * 64 + T - 1) / T;
  const int gEW   = (E * 32 + T - 1) / T;             // wave-per-edge
  const int gX4   = (N * 128 / 4 + T - 1) / T;
  const int gH4   = (N * 64 / 4 + T - 1) / T;

  // D^{-1/2}
  deg_init_kernel<<<gN, T, 0, stream>>>(dinv, N);
  deg_count_kernel<<<gE, T, 0, stream>>>(dst, dinv, E);
  deg_rsqrt_kernel<<<gN, T, 0, stream>>>(dinv, N);

  // operand preparation (bf16 conversion + weight swizzle)
  f32_to_bf16_kernel<<<gX4, T, 0, stream>>>(x, xbf, N * 128 / 4);
  weight_swizzle_kernel<<<(128 * 64 + T - 1) / T, T, 0, stream>>>(W1, wbf1, 128 * 64);
  weight_swizzle_kernel<<<(64 * 64 + T - 1) / T, T, 0, stream>>>(W2, wbf2, 64 * 64);

  // conv1
  gemm_bf16_wmma_kernel<128><<<gGemm, T, 0, stream>>>(xbf, wbf1, buf1, m_tiles);
  acc_init_kernel<<<gNC, T, 0, stream>>>(buf1, dinv, b1, buf2, N);   // overwrites xbf (dead)
  scatter_edges_kernel<<<gEW, T, 0, stream>>>(src, dst, dinv, buf1, buf2, E);
  tanh_to_bf16_kernel<<<gH4, T, 0, stream>>>(buf2, hbf, N * 64 / 4);

  // conv2
  gemm_bf16_wmma_kernel<64><<<gGemm, T, 0, stream>>>(hbf, wbf2, buf1, m_tiles);
  acc_init_kernel<<<gNC, T, 0, stream>>>(buf1, dinv, b2, buf2, N);
  scatter_edges_kernel<<<gEW, T, 0, stream>>>(src, dst, dinv, buf1, buf2, E);

  // head (fuses tanh of conv2 output)
  mlp_head_kernel<<<gN, T, 0, stream>>>(buf2, fw1, fb1, fw2, fb2, out, N);
}
